// Encoder_Decoder_Quaternion_86122684219687
// MI455X (gfx1250) — matrix operand order repaired and checked by reading, not compiled, not run on hardware
//
#include <hip/hip_runtime.h>
#include <hip/hip_bf16.h>

// ---------------------------------------------------------------------------
// Quaternion RNN encoder-decoder for MI455X (gfx1250, wave32).
//   * Quaternion linears expanded once to dense f16 matrices (N-major):
//       Wcat_t = [Wx|Wh] (1024 x 1120), Wl_t (96 x 1024).
//   * RNN step = one WMMA GEMM [512 x 1120]x[1120 x 1024] -> tanh -> f16
//     written into the h-region of the next step's activation buffer.
//   * Predict head = WMMA GEMM [512 x 1024]x[1024 x 96] -> f32 + pointwise
//     quaternion normalize / Hamilton product; decoder feeds pred back (f16).
//   * v_wmma_f32_16x16x32_f16 (f16 in, f32 accum): activations tanh-bounded,
//     weights ~0.05 -> f16 safe; 8x the f32 16x16x4 WMMA rate.
//   * 2-stage software pipeline over double-buffered LDS: the ds_store that
//     consumes a global load happens one full iteration after its issue, so
//     weight-fetch latency hides under the WMMAs (serial RNN => latency-bound).
// ---------------------------------------------------------------------------

typedef __attribute__((ext_vector_type(16))) _Float16 v16h;
typedef __attribute__((ext_vector_type(8)))  float    v8f;

#define T_ENC 100
#define T_DEC 25
#define BATCH 512
#define DIN   96
#define HID   1024
#define KCAT  1120          // 96 (x) + 1024 (h)
#define LDA   1120
#define LP    40            // LDS pitch in halves (80B) -> conflict-free b128

__device__ __constant__ int   qcomp[4][4] = {{0,1,2,3},{1,0,3,2},{2,3,0,1},{3,2,1,0}};
__device__ __constant__ float qsign[4][4] = {{ 1.f, 1.f, 1.f, 1.f},
                                             {-1.f, 1.f,-1.f, 1.f},
                                             {-1.f, 1.f, 1.f,-1.f},
                                             {-1.f,-1.f, 1.f, 1.f}};

// ---- weight expansion -----------------------------------------------------
__global__ void build_wcat_kernel(const float* __restrict__ Wx,   // [4,24,256]
                                  const float* __restrict__ Wh,   // [4,256,256]
                                  _Float16* __restrict__ Wcat) {  // [1024,1120]
  int idx = blockIdx.x * blockDim.x + threadIdx.x;
  if (idx >= HID * KCAT) return;
  int n = idx / KCAT, k = idx % KCAT;
  int co = n >> 8, oc = n & 255;
  float v;
  if (k < DIN) {
    int ci = k / 24, ic = k % 24;
    v = qsign[ci][co] * Wx[qcomp[ci][co] * (24 * 256) + ic * 256 + oc];
  } else {
    int kh = k - DIN;
    int ci = kh >> 8, ic = kh & 255;
    v = qsign[ci][co] * Wh[qcomp[ci][co] * (256 * 256) + ic * 256 + oc];
  }
  Wcat[idx] = (_Float16)v;
}

__global__ void build_wl_kernel(const float* __restrict__ Wl,   // [4,256,24]
                                _Float16* __restrict__ Wlt) {   // [96,1024]
  int idx = blockIdx.x * blockDim.x + threadIdx.x;
  if (idx >= DIN * HID) return;
  int n = idx / HID, k = idx % HID;
  int co = n / 24, oc = n % 24;
  int ci = k >> 8, ic = k & 255;
  Wlt[idx] = (_Float16)(qsign[ci][co] * Wl[qcomp[ci][co] * (256 * 24) + ic * 24 + oc]);
}

// ---- activation buffer maintenance ---------------------------------------
__global__ void zero_h_kernel(_Float16* __restrict__ abuf) {
  int i = blockIdx.x * blockDim.x + threadIdx.x;
  if (i >= BATCH * HID) return;
  int b = i >> 10, c = i & 1023;
  abuf[b * LDA + DIN + c] = (_Float16)0.0f;
}

__global__ void setx_kernel(const float* __restrict__ src,      // [512,96] f32
                            _Float16* __restrict__ abuf) {      // x region
  int i = blockIdx.x * blockDim.x + threadIdx.x;
  if (i >= BATCH * DIN) return;
  int b = i / DIN, c = i % DIN;
  abuf[b * LDA + c] = (_Float16)src[i];
}

// ---- WMMA GEMM: C[M,N] = act(A[M,K] @ Bt[N,K]^T + bias) -------------------
// 256 threads = 8 waves; WG tile 64(M) x 128(N); wave tile 32x32 (4 accum).
// Double-buffered LDS; single barrier per 32-wide K slab; loads issued one
// full iteration before their consuming ds_store.
template <bool TANH_TO_F16>
__global__ __launch_bounds__(256)
void wmma_gemm_kernel(const _Float16* __restrict__ A, int lda,
                      const _Float16* __restrict__ Bt, int ldb,
                      const float* __restrict__ bias,
                      void* __restrict__ Cout, int ldc,
                      int N, int K) {
  __shared__ _Float16 As[2][64 * LP];
  __shared__ _Float16 Bs[2][128 * LP];

  const int tid  = threadIdx.x;
  const int lane = tid & 31;
  const int wave = tid >> 5;
  const int wm   = wave >> 2;           // 0..1  (M sub-block of 32)
  const int wn   = wave & 3;            // 0..3  (N sub-block of 32)
  const int rowBlk = blockIdx.y * 64;
  const int colBlk = blockIdx.x * 128;

  const int l15 = lane & 15;
  const int lhi = lane >> 4;

  // staging assignments (per thread): A one b128, B two b128
  const int arow  = tid >> 2;           // 0..63
  const int akoff = (tid & 3) * 8;      // 0,8,16,24 halves
  const int brow  = tid >> 1;           // 0..127
  const int bkoff = (tid & 1) * 16;     // 0,16 halves
  int bsrc = colBlk + brow; if (bsrc >= N) bsrc = N - 1;   // clamp (N=96 tail)

  const _Float16* aGp = A + (size_t)(rowBlk + arow) * lda + akoff;
  const _Float16* bGp = Bt + (size_t)bsrc * ldb + bkoff;

  // pull the weight tile rows toward this WGP while we set up
  __builtin_prefetch(bGp, 0, 3);
  __builtin_prefetch(bGp + 64 * (size_t)ldb, 0, 3);

  const int aso = arow * LP + akoff;
  const int bso = brow * LP + bkoff;

  v8f acc[2][2] = {};
  const int kTiles = K >> 5;

  // prologue: stage slab 0 into buffer 0; issue loads for slab 1
  {
    uint4 ra  = *(const uint4*)(aGp);
    uint4 rb0 = *(const uint4*)(bGp);
    uint4 rb1 = *(const uint4*)(bGp + 8);
    *(uint4*)&As[0][aso] = ra;
    *(uint4*)&Bs[0][bso] = rb0;
    *(uint4*)&Bs[0][bso + 8] = rb1;
  }
  uint4 ra, rb0, rb1;
  if (kTiles > 1) {
    ra  = *(const uint4*)(aGp + 32);
    rb0 = *(const uint4*)(bGp + 32);
    rb1 = *(const uint4*)(bGp + 40);
  }

  for (int kt = 0; kt < kTiles; ++kt) {
    const int cur = kt & 1;
    __syncthreads();   // buf[cur] staged; all waves done READING buf[cur^1]

    // consume last iteration's global loads: stage slab kt+1 now, at the TOP,
    // so its s_wait_loadcnt lands a full iteration after the load was issued.
    if (kt + 1 < kTiles) {
      *(uint4*)&As[cur ^ 1][aso] = ra;
      *(uint4*)&Bs[cur ^ 1][bso] = rb0;
      *(uint4*)&Bs[cur ^ 1][bso + 8] = rb1;
    }
    // issue global loads for slab kt+2; they fly under the frag loads,
    // 4 WMMAs and the next barrier before being consumed.
    if (kt + 2 < kTiles) {
      const int kg = (kt + 2) << 5;
      ra  = *(const uint4*)(aGp + kg);
      rb0 = *(const uint4*)(bGp + kg);
      rb1 = *(const uint4*)(bGp + kg + 8);
    }

    const _Float16* as = As[cur];
    const _Float16* bs = Bs[cur];

    // A fragment (ISA 16-bit A 16x32 layout): lanes 0-15 -> K base 0,
    // lanes 16-31 -> K base 8; halves 0..7 = K kb..kb+7, 8..15 = kb+16..+23.
    union { v16h h; uint4 u[2]; } a0, a1, b0, b1;
    {
      const int kb = lhi * 8;
      const int r0 = (wm * 32 + l15) * LP;
      a0.u[0] = *(const uint4*)&as[r0 + kb];
      a0.u[1] = *(const uint4*)&as[r0 + kb + 16];
      const int r1 = r0 + 16 * LP;
      a1.u[0] = *(const uint4*)&as[r1 + kb];
      a1.u[1] = *(const uint4*)&as[r1 + kb + 16];
    }
    // B fragment (32x16), same K order as A: lanes 0-15 -> K base 0,
    // lanes 16-31 -> K base 8; halves 0..7 = K kb..kb+7, 8..15 = kb+16..+23.
    {
      const int kb2 = lhi * 8;
      const int c0 = (wn * 32 + l15) * LP;
      b0.u[0] = *(const uint4*)&bs[c0 + kb2];
      b0.u[1] = *(const uint4*)&bs[c0 + kb2 + 16];
      const int c1 = c0 + 16 * LP;
      b1.u[0] = *(const uint4*)&bs[c1 + kb2];
      b1.u[1] = *(const uint4*)&bs[c1 + kb2 + 16];
    }

    acc[0][0] = __builtin_amdgcn_wmma_f32_16x16x32_f16(false, a0.h, false, b0.h,
                                                       (short)0, acc[0][0], false, false);
    acc[0][1] = __builtin_amdgcn_wmma_f32_16x16x32_f16(false, a0.h, false, b1.h,
                                                       (short)0, acc[0][1], false, false);
    acc[1][0] = __builtin_amdgcn_wmma_f32_16x16x32_f16(false, a1.h, false, b0.h,
                                                       (short)0, acc[1][0], false, false);
    acc[1][1] = __builtin_amdgcn_wmma_f32_16x16x32_f16(false, a1.h, false, b1.h,
                                                       (short)0, acc[1][1], false, false);
  }

  // Epilogue. C tile layout: col = l15, rows = lhi*8 + vgpr index.
  for (int mi = 0; mi < 2; ++mi) {
    for (int ni = 0; ni < 2; ++ni) {
      const int col = colBlk + wn * 32 + ni * 16 + l15;
      if (col >= N) continue;
      const float bv = bias[col];
      const int rowBase = rowBlk + wm * 32 + mi * 16 + lhi * 8;
#pragma unroll
      for (int r = 0; r < 8; ++r) {
        const float v = acc[mi][ni][r] + bv;
        if (TANH_TO_F16) {
          ((_Float16*)Cout)[(size_t)(rowBase + r) * ldc + col] = (_Float16)tanhf(v);
        } else {
          ((float*)Cout)[(size_t)(rowBase + r) * ldc + col] = v;
        }
      }
    }
  }
}

// ---- predict pointwise: normalize quaternions + Hamilton product ----------
__global__ void pw_predict_kernel(const float* __restrict__ P,     // [512,96]
                                  const float* __restrict__ pre,   // [512,96]
                                  float* __restrict__ out,         // [512,96]
                                  _Float16* __restrict__ xnext) {  // abuf or null
  int idx = blockIdx.x * blockDim.x + threadIdx.x;
  if (idx >= BATCH * 24) return;
  int b = idx / 24, c = idx % 24;
  const float* pb = P + b * DIN;
  float w = pb[c], x = pb[24 + c], y = pb[48 + c], z = pb[72 + c];
  float nrm = sqrtf(w * w + x * x + y * y + z * z);
  float inv = 1.0f / fmaxf(nrm, 1e-12f);
  w *= inv; x *= inv; y *= inv; z *= inv;
  const float* qb = pre + b * DIN;
  float rw = qb[c], rx = qb[24 + c], ry = qb[48 + c], rz = qb[72 + c];
  float ow = w * rw - x * rx - y * ry - z * rz;
  float ox = w * rx + x * rw + y * rz - z * ry;
  float oy = w * ry - x * rz + y * rw + z * rx;
  float oz = w * rz + x * ry - y * rx + z * rw;
  float* ob = out + b * DIN;
  ob[c] = ow; ob[24 + c] = ox; ob[48 + c] = oy; ob[72 + c] = oz;
  if (xnext) {
    _Float16* xb = xnext + b * LDA;
    xb[c]      = (_Float16)ow;
    xb[24 + c] = (_Float16)ox;
    xb[48 + c] = (_Float16)oy;
    xb[72 + c] = (_Float16)oz;
  }
}

// ---------------------------------------------------------------------------
extern "C" void kernel_launch(void* const* d_in, const int* in_sizes, int n_in,
                              void* d_out, int out_size, void* d_ws, size_t ws_size,
                              hipStream_t stream) {
  const float* inp    = (const float*)d_in[0];  // [100,512,96]
  const float* target = (const float*)d_in[1];  // [26,512,96]
  const float* Wx     = (const float*)d_in[2];  // [4,24,256]
  const float* bx     = (const float*)d_in[3];  // [1024]
  const float* Wh     = (const float*)d_in[4];  // [4,256,256]
  const float* Wl     = (const float*)d_in[5];  // [4,256,24]
  const float* bl     = (const float*)d_in[6];  // [96]
  float* out = (float*)d_out;

  // workspace layout (all 256B aligned)
  char* ws = (char*)d_ws;
  _Float16* wcat  = (_Float16*)(ws);                       // 1024*1120*2 = 2293760
  _Float16* wlt   = (_Float16*)(ws + 2293760);             //   96*1024*2 =  196608
  _Float16* abuf0 = (_Float16*)(ws + 2490368);             //  512*1120*2 = 1146880
  _Float16* abuf1 = (_Float16*)(ws + 3637248);             //  512*1120*2
  float*    pbuf  = (float*)   (ws + 4784128);             //   512*96*4  =  196608
  _Float16* abuf[2] = {abuf0, abuf1};

  const int STEP = BATCH * DIN;   // 49152 floats per timestep slab

  // weight expansion (every call: deterministic, ws is not preserved)
  build_wcat_kernel<<<(HID * KCAT + 255) / 256, 256, 0, stream>>>(Wx, Wh, wcat);
  build_wl_kernel<<<(DIN * HID + 255) / 256, 256, 0, stream>>>(Wl, wlt);

  // h0 = 0, x0 = inp[0]
  zero_h_kernel<<<(BATCH * HID + 255) / 256, 256, 0, stream>>>(abuf[0]);
  setx_kernel<<<(STEP + 255) / 256, 256, 0, stream>>>(inp, abuf[0]);

  const dim3 gRnn(HID / 128, BATCH / 64);   // 8 x 8
  const dim3 gPred(1, BATCH / 64);          // 1 x 8

  int cur = 0;
  // ---- encoder: 100 steps, predict fused per step (avoids storing hs) ----
  for (int t = 0; t < T_ENC; ++t) {
    const int nxt = cur ^ 1;
    wmma_gemm_kernel<true><<<gRnn, 256, 0, stream>>>(
        abuf[cur], LDA, wcat, KCAT, bx, (void*)(abuf[nxt] + DIN), LDA, HID, KCAT);
    // x for the next RNN step: inp[t+1], or target[0] to prime the decoder
    const float* xsrc = (t < T_ENC - 1) ? (inp + (size_t)(t + 1) * STEP) : target;
    setx_kernel<<<(STEP + 255) / 256, 256, 0, stream>>>(xsrc, abuf[nxt]);
    wmma_gemm_kernel<false><<<gPred, 256, 0, stream>>>(
        abuf[nxt] + DIN, LDA, wlt, HID, bl, (void*)pbuf, DIN, DIN, HID);
    pw_predict_kernel<<<(BATCH * 24 + 255) / 256, 256, 0, stream>>>(
        pbuf, inp + (size_t)t * STEP, out + (size_t)t * STEP, nullptr);
    cur = nxt;
  }

  // ---- decoder: 25 steps, prediction fed back as next input -------------
  for (int t = 0; t < T_DEC; ++t) {
    const int nxt = cur ^ 1;
    wmma_gemm_kernel<true><<<gRnn, 256, 0, stream>>>(
        abuf[cur], LDA, wcat, KCAT, bx, (void*)(abuf[nxt] + DIN), LDA, HID, KCAT);
    wmma_gemm_kernel<false><<<gPred, 256, 0, stream>>>(
        abuf[nxt] + DIN, LDA, wlt, HID, bl, (void*)pbuf, DIN, DIN, HID);
    const float* pre = (t == 0) ? target
                                : out + (size_t)(T_ENC + t - 1) * STEP;
    pw_predict_kernel<<<(BATCH * 24 + 255) / 256, 256, 0, stream>>>(
        pbuf, pre, out + (size_t)(T_ENC + t) * STEP, abuf[nxt]);
    cur = nxt;
  }
}
